// AIR_8220567405312
// MI455X (gfx1250) — compile-verified, matrix operand order checked by reading, not run on hardware
//
#include <hip/hip_runtime.h>
#include <hip/hip_bf16.h>
#include <math.h>

// ---------------- problem constants ----------------
#define Bn   4096
#define Tn   3
#define LHn  256
#define NGn  1024      // 4*LH gate width
#define KXp  2560      // x K (2500) padded to mult of 32
#define KZp  320       // [z(54)->64 pad] + h(256)
#define KOp  416       // obj K (400) padded
#define NE1p 256       // enc hidden (200) padded to mult 64 (doubles as K of enc2)
#define NE2p 128       // enc out (100) padded to mult 64
#define KWp  64        // z_what K (50) padded
#define ND1p 256       // dec hidden (200) padded to mult 64 (doubles as K of dec2)
#define ND2p 2560      // dec out (2500) padded to mult 64

typedef __attribute__((ext_vector_type(16))) _Float16 v16h;
typedef __attribute__((ext_vector_type(8)))  _Float16 v8h;
typedef __attribute__((ext_vector_type(8)))  float    v8f;

__device__ __forceinline__ float sigmoidf_(float x) { return 1.f / (1.f + expf(-x)); }
__device__ __forceinline__ float softplusf_(float x) { return x > 20.f ? x : log1pf(expf(x)); }
__device__ __forceinline__ float nkl_(float mu, float sg, float mu0, float sg0) {
    float d = mu - mu0;
    return logf(sg0 / sg) + (sg * sg + d * d) / (2.f * sg0 * sg0) - 0.5f;
}
// bilinear corner gather with border validity (matches reference gather())
__device__ __forceinline__ float bsamp_(const float* im, float xi, float yi, int Wn, int Hn) {
    bool valid = (xi >= 0.f) && (xi <= (float)(Wn - 1)) && (yi >= 0.f) && (yi <= (float)(Hn - 1));
    int xc = (int)fminf(fmaxf(xi, 0.f), (float)(Wn - 1));
    int yc = (int)fminf(fmaxf(yi, 0.f), (float)(Hn - 1));
    float v = im[yc * Wn + xc];
    return valid ? v : 0.f;
}

// A-fragment (and B-fragment) per-lane layout for v_wmma_f32_16x16x32_f16:
// lane holds row/col (lane&15); its 16 halves are two contiguous 16B runs at
// K offsets (lane>>4)*8 and (lane>>4)*8 + 16.
__device__ __forceinline__ v16h ldfrag_(const _Float16* __restrict__ p) {
    v8h lo = *(const v8h*)p;
    v8h hi = *(const v8h*)(p + 16);
    return __builtin_shufflevector(lo, hi, 0,1,2,3,4,5,6,7,8,9,10,11,12,13,14,15);
}

// ---------------- WMMA GEMM: out[M x N] = act(A[M x K] * W[N x K]^T + Cadd + bias) ----------
// 2x2 register blocking: each wave owns a 32x32 output tile (4 accumulators),
// 4 WMMAs per 2 A-frag + 2 B-frag loads -> 2 b128 loads per WMMA.
// Block = 256 thr (8 waves, 4x2) -> 128x64 block tile. grid = (N/64, M/128).
__global__ void wmma_gemm_f16(const _Float16* __restrict__ A, int lda,
                              const _Float16* __restrict__ Wt, int ldw,
                              void* __restrict__ out, int ldo,
                              const float* __restrict__ Cadd, int ldc,
                              const float* __restrict__ bias,
                              int K, int act, int outF16) {
    const int lane = threadIdx.x & 31;
    const int wv   = threadIdx.x >> 5;                 // 0..7
    const int m0   = blockIdx.y * 128 + (wv >> 1) * 32;
    const int n0   = blockIdx.x * 64  + (wv & 1) * 32;
    const int r    = lane & 15;
    const int hb   = (lane >> 4) * 8;                  // K sub-base for this half-wave

    const _Float16* pa0 = A  + (size_t)(m0 + r) * lda + hb;
    const _Float16* pa1 = pa0 + (size_t)16 * lda;
    const _Float16* pb0 = Wt + (size_t)(n0 + r) * ldw + hb;
    const _Float16* pb1 = pb0 + (size_t)16 * ldw;

    v8f acc00 = {}, acc01 = {}, acc10 = {}, acc11 = {};
    for (int k = 0; k < K; k += 32) {
        __builtin_prefetch((const void*)(pa0 + k + 64), 0, 1);
        __builtin_prefetch((const void*)(pb0 + k + 64), 0, 1);
        v16h a0 = ldfrag_(pa0 + k);
        v16h a1 = ldfrag_(pa1 + k);
        v16h b0 = ldfrag_(pb0 + k);
        v16h b1 = ldfrag_(pb1 + k);
        acc00 = __builtin_amdgcn_wmma_f32_16x16x32_f16(false, a0, false, b0, (short)0, acc00, false, false);
        acc01 = __builtin_amdgcn_wmma_f32_16x16x32_f16(false, a0, false, b1, (short)0, acc01, false, false);
        acc10 = __builtin_amdgcn_wmma_f32_16x16x32_f16(false, a1, false, b0, (short)0, acc10, false, false);
        acc11 = __builtin_amdgcn_wmma_f32_16x16x32_f16(false, a1, false, b1, (short)0, acc11, false, false);
    }

    // C/D layout: VGPR i -> M = tile_m + i + (lane>=16)*8 ; N = tile_n + (lane&15)
    const int ro = (lane >> 4) * 8;
    const int co = lane & 15;
    auto store_tile = [&](v8f acc, int tm, int tn) {
#pragma unroll
        for (int i = 0; i < 8; ++i) {
            float v = acc[i];
            int row = tm + ro + i;
            int col = tn + co;
            if (Cadd) v += Cadd[(size_t)row * ldc + col];
            if (bias) v += bias[col];
            if (act == 1) v = v > 0.f ? v : 0.f;            // relu
            else if (act == 2) v = sigmoidf_(v - 2.f);      // sigmoid(x-2)
            if (outF16) ((_Float16*)out)[(size_t)row * ldo + col] = (_Float16)v;
            else        ((float*)out)[(size_t)row * ldo + col] = v;
        }
    };
    store_tile(acc00, m0,      n0);
    store_tile(acc01, m0,      n0 + 16);
    store_tile(acc10, m0 + 16, n0);
    store_tile(acc11, m0 + 16, n0 + 16);
}

// ---------------- staging / helper kernels ----------------
__global__ void cvt_pad_f16_k(_Float16* __restrict__ dst, const float* __restrict__ src,
                              int Rs, int Cs, int srcLd, int Cd, long total) {
    long i = (long)blockIdx.x * 256 + threadIdx.x;
    if (i >= total) return;
    int r = (int)(i / Cd), c = (int)(i % Cd);
    dst[i] = (r < Rs && c < Cs) ? (_Float16)src[(size_t)r * srcLd + c] : (_Float16)0.f;
}

__global__ void build_w2_k(_Float16* __restrict__ dst, const float* __restrict__ Wih,
                           const float* __restrict__ Whh) {
    long i = (long)blockIdx.x * 256 + threadIdx.x;
    if (i >= (long)NGn * KZp) return;
    int n = (int)(i / KZp), c = (int)(i % KZp);
    float v;
    if (c < 54)      v = Wih[(size_t)n * 2554 + 2500 + c];  // z_where(3)+z_what(50)+z_pres(1)
    else if (c < 64) v = 0.f;
    else             v = Whh[(size_t)n * 256 + (c - 64)];
    dst[i] = (_Float16)v;
}

__global__ void add2_k(float* dst, const float* a, const float* b, int n) {
    int i = blockIdx.x * 256 + threadIdx.x;
    if (i < n) dst[i] = a[i] + b[i];
}

__global__ void pad_bias_k(float* dst, const float* src, int Cs, int Cd) {
    int i = blockIdx.x * 256 + threadIdx.x;
    if (i < Cd) dst[i] = (i < Cs) ? src[i] : 0.f;
}

__global__ void init_state_k(float* h, float* c, float* blh, float* blc, float* canvas,
                             float* zpres, float* zwhere, float* zwhat,
                             float* kl, float* bv, float* zpl, float* mask) {
    long i = (long)blockIdx.x * 256 + threadIdx.x;
    if (i < (long)Bn * 2500) canvas[i] = 0.f;
    if (i < (long)Bn * LHn) { h[i] = 0.f; c[i] = 0.f; blh[i] = 0.f; blc[i] = 0.f; }
    if (i < Bn)      zpres[i] = 1.f;
    if (i < Bn * 3)  zwhere[i] = 0.f;
    if (i < Bn * 50) zwhat[i] = 0.f;
    if (i < Bn * Tn) { kl[i] = 0.f; bv[i] = 0.f; zpl[i] = 0.f; mask[i] = 0.f; }
}

__global__ void build_zh_k(_Float16* __restrict__ zh, const float* zw, const float* zwhat,
                           const float* zp, const float* h) {
    long i = (long)blockIdx.x * 256 + threadIdx.x;
    if (i >= (long)Bn * KZp) return;
    int b = (int)(i / KZp), c = (int)(i % KZp);
    float v;
    if (c < 3)       v = zw[b * 3 + c];
    else if (c < 53) v = zwhat[b * 50 + (c - 3)];
    else if (c == 53) v = zp[b];
    else if (c < 64) v = 0.f;
    else             v = h[(size_t)b * LHn + (c - 64)];
    zh[i] = (_Float16)v;
}

__global__ void lstm_gate_k(const float* __restrict__ gates, float* __restrict__ h,
                            float* __restrict__ c) {
    long i = (long)blockIdx.x * 256 + threadIdx.x;
    if (i >= (long)Bn * LHn) return;
    int b = (int)(i / LHn), j = (int)(i % LHn);
    const float* g = gates + (size_t)b * NGn;
    float gi = g[j], gf = g[256 + j], gg = g[512 + j], go = g[768 + j];
    float cc = sigmoidf_(gf) * c[i] + sigmoidf_(gi) * tanhf(gg);
    c[i] = cc;
    h[i] = sigmoidf_(go) * tanhf(cc);
}

// fused predictor: z = h@W_p^T + b_p, pres sample, z_where sample, kl_pres+kl_where, zpl, mask
__global__ void predictor_k(const float* __restrict__ h, const float* __restrict__ Wp,
                            const float* __restrict__ bp, const float* __restrict__ u_pres,
                            const float* __restrict__ eps_w, float* __restrict__ zpres,
                            float* __restrict__ zwhere, float* __restrict__ zpl,
                            float* __restrict__ mask, float* __restrict__ kl, int t) {
    int b = blockIdx.x * 256 + threadIdx.x;
    if (b >= Bn) return;
    const float* hb = h + (size_t)b * LHn;
    float z[7];
#pragma unroll
    for (int j = 0; j < 7; ++j) {
        float s = bp[j];
        const float* w = Wp + j * LHn;
        for (int k = 0; k < LHn; ++k) s += hb[k] * w[k];
        z[j] = s;
    }
    float p = fminf(fmaxf(sigmoidf_(z[0]), 1e-6f), 1.f - 1e-6f);
    float samp = (u_pres[b * Tn + t] < p) ? 1.f : 0.f;
    float zp = samp * zpres[b];
    zpres[b] = zp;
    float logprob = zp * logf(p) + (1.f - zp) * logf(1.f - p);
    zpl[b * Tn + t]  = logprob * zp;
    mask[b * Tn + t] = zp;
    float klacc = p * logf(2.f * p) + (1.f - p) * logf(2.f * (1.f - p));
    const float wploc[3] = {3.f, 0.f, 0.f};
    const float wpscl[3] = {0.2f, 1.f, 1.f};
#pragma unroll
    for (int d = 0; d < 3; ++d) {
        float loc = z[1 + d], scl = softplusf_(z[4 + d]);
        zwhere[b * 3 + d] = loc + scl * eps_w[(size_t)(b * Tn + t) * 3 + d];
        klacc += nkl_(loc, scl, wploc[d], wpscl[d]);
    }
    kl[b * Tn + t] = klacc;   // kl_what added later
}

__global__ void baseline_out_k(const float* __restrict__ blh, const float* __restrict__ Wbl,
                               const float* __restrict__ bbl, const float* __restrict__ zpres,
                               float* __restrict__ bv, int t) {
    int b = blockIdx.x * 256 + threadIdx.x;
    if (b >= Bn) return;
    const float* hb = blh + (size_t)b * LHn;
    float s = bbl[0];
    for (int k = 0; k < LHn; ++k) s += hb[k] * Wbl[k];
    bv[b * Tn + t] = s * zpres[b];
}

// attention glimpse: inverse STN, 20x20 bilinear sample from x(50x50), write padded f16
__global__ void obj_sample_k(const float* __restrict__ x, const float* __restrict__ zwhere,
                             _Float16* __restrict__ obj16) {
    long i = (long)blockIdx.x * 256 + threadIdx.x;
    if (i >= (long)Bn * KOp) return;
    int b = (int)(i / KOp), c = (int)(i % KOp);
    if (c >= 400) { obj16[i] = (_Float16)0.f; return; }
    int rr = c / 20, cc = c % 20;
    float gx = -1.f + 2.f * cc / 19.f, gy = -1.f + 2.f * rr / 19.f;
    float s = zwhere[b * 3], tx = zwhere[b * 3 + 1], ty = zwhere[b * 3 + 2];
    float si = 1.f / s, txi = -tx / s, tyi = -ty / s;
    float u = si * gx + txi, v = si * gy + tyi;
    float xf = (u + 1.f) * 0.5f * 49.f, yf = (v + 1.f) * 0.5f * 49.f;
    float x0 = floorf(xf), y0 = floorf(yf), wx = xf - x0, wy = yf - y0;
    const float* im = x + (size_t)b * 2500;
    float val = (1.f - wx) * (1.f - wy) * bsamp_(im, x0,       y0,       50, 50)
              + wx         * (1.f - wy) * bsamp_(im, x0 + 1.f, y0,       50, 50)
              + (1.f - wx) * wy         * bsamp_(im, x0,       y0 + 1.f, 50, 50)
              + wx         * wy         * bsamp_(im, x0 + 1.f, y0 + 1.f, 50, 50);
    obj16[i] = (_Float16)val;
}

// z_what sampling + kl_what accumulate; also writes f16 padded decoder input
__global__ void zwhat_k(const float* __restrict__ e2, const float* __restrict__ eps_what,
                        float* __restrict__ zwhat, _Float16* __restrict__ zw16,
                        float* __restrict__ kl, int t) {
    int b = blockIdx.x * 256 + threadIdx.x;
    if (b >= Bn) return;
    const float* e = e2 + (size_t)b * NE2p;
    float klw = 0.f;
    for (int d = 0; d < 50; ++d) {
        float loc = e[d], scl = softplusf_(e[50 + d]);
        float zv = loc + scl * eps_what[(size_t)(b * Tn + t) * 50 + d];
        zwhat[b * 50 + d] = zv;
        zw16[(size_t)b * KWp + d] = (_Float16)zv;
        klw += nkl_(loc, scl, 0.f, 1.f);
    }
    for (int d = 50; d < KWp; ++d) zw16[(size_t)b * KWp + d] = (_Float16)0.f;
    kl[b * Tn + t] += klw;
}

// render: forward STN of dec(50x50) onto 50x50, canvas += img * z_pres
__global__ void render_k(const float* __restrict__ dec, const float* __restrict__ zwhere,
                         const float* __restrict__ zpres, float* __restrict__ canvas) {
    long i = (long)blockIdx.x * 256 + threadIdx.x;
    if (i >= (long)Bn * 2500) return;
    int b = (int)(i / 2500), pix = (int)(i % 2500);
    int rr = pix / 50, cc = pix % 50;
    float gx = -1.f + 2.f * cc / 49.f, gy = -1.f + 2.f * rr / 49.f;
    float s = zwhere[b * 3], tx = zwhere[b * 3 + 1], ty = zwhere[b * 3 + 2];
    float u = s * gx + tx, v = s * gy + ty;
    float xf = (u + 1.f) * 0.5f * 49.f, yf = (v + 1.f) * 0.5f * 49.f;
    float x0 = floorf(xf), y0 = floorf(yf), wx = xf - x0, wy = yf - y0;
    const float* im = dec + (size_t)b * ND2p;   // 50-wide rows within first 2500 cols
    float val = (1.f - wx) * (1.f - wy) * bsamp_(im, x0,       y0,       50, 50)
              + wx         * (1.f - wy) * bsamp_(im, x0 + 1.f, y0,       50, 50)
              + (1.f - wx) * wy         * bsamp_(im, x0,       y0 + 1.f, 50, 50)
              + wx         * wy         * bsamp_(im, x0 + 1.f, y0 + 1.f, 50, 50);
    canvas[i] += val * zpres[b];
}

// per-image Gaussian log-likelihood sum: block per b
__global__ void ll_k(const float* __restrict__ x, const float* __restrict__ canvas,
                     float* __restrict__ ll) {
    __shared__ float red[256];
    int b = blockIdx.x;
    float s = 0.f;
    for (int i = threadIdx.x; i < 2500; i += 256) {
        float d = x[(size_t)b * 2500 + i] - canvas[(size_t)b * 2500 + i];
        // -0.5*(d/0.3)^2 - log(0.3) - 0.5*log(2*pi)
        s += -d * d * 5.5555553f + 0.28503427f;
    }
    red[threadIdx.x] = s;
    __syncthreads();
    for (int st = 128; st > 0; st >>= 1) {
        if (threadIdx.x < st) red[threadIdx.x] += red[threadIdx.x + st];
        __syncthreads();
    }
    if (threadIdx.x == 0) ll[b] = red[0];
}

// final scalar: loss.mean + baseline_loss.mean
__global__ void final_k(const float* __restrict__ kl, const float* __restrict__ bv,
                        const float* __restrict__ zpl, const float* __restrict__ mask,
                        const float* __restrict__ ll, float* __restrict__ out) {
    __shared__ float r1[256], r2[256];
    float lsum = 0.f, bsum = 0.f;
    for (int b = threadIdx.x; b < Bn; b += 256) {
        float k0 = kl[b * 3], k1 = kl[b * 3 + 1], k2 = kl[b * 3 + 2];
        float l = ll[b];
        float ls0 = (k1 + k2 + l) * mask[b * 3];
        float ls1 = (k2 + l) * mask[b * 3 + 1];
        float ls2 = (l) * mask[b * 3 + 2];
        float reinf = (ls0 - bv[b * 3]) * zpl[b * 3]
                    + (ls1 - bv[b * 3 + 1]) * zpl[b * 3 + 1]
                    + (ls2 - bv[b * 3 + 2]) * zpl[b * 3 + 2];
        lsum += reinf + (k0 + k1 + k2) - l;
        float d0 = bv[b * 3] - ls0, d1 = bv[b * 3 + 1] - ls1, d2 = bv[b * 3 + 2] - ls2;
        bsum += d0 * d0 + d1 * d1 + d2 * d2;
    }
    r1[threadIdx.x] = lsum; r2[threadIdx.x] = bsum;
    __syncthreads();
    for (int st = 128; st > 0; st >>= 1) {
        if (threadIdx.x < st) { r1[threadIdx.x] += r1[threadIdx.x + st]; r2[threadIdx.x] += r2[threadIdx.x + st]; }
        __syncthreads();
    }
    if (threadIdx.x == 0) out[0] = r1[0] / (float)Bn + r2[0] / (float)(Bn * Tn);
}

// ---------------- host driver ----------------
static inline int gb(long n) { return (int)((n + 255) / 256); }

extern "C" void kernel_launch(void* const* d_in, const int* in_sizes, int n_in,
                              void* d_out, int out_size, void* d_ws, size_t ws_size,
                              hipStream_t stream) {
    const float* x      = (const float*)d_in[0];
    const float* W_ih   = (const float*)d_in[1];
    const float* W_hh   = (const float*)d_in[2];
    const float* b_ih   = (const float*)d_in[3];
    const float* b_hh   = (const float*)d_in[4];
    const float* W_p    = (const float*)d_in[5];
    const float* b_p    = (const float*)d_in[6];
    const float* W_e1   = (const float*)d_in[7];
    const float* b_e1   = (const float*)d_in[8];
    const float* W_e2   = (const float*)d_in[9];
    const float* b_e2   = (const float*)d_in[10];
    const float* W_d1   = (const float*)d_in[11];
    const float* b_d1   = (const float*)d_in[12];
    const float* W_d2   = (const float*)d_in[13];
    const float* b_d2   = (const float*)d_in[14];
    const float* bW_ih  = (const float*)d_in[15];
    const float* bW_hh  = (const float*)d_in[16];
    const float* bb_ih  = (const float*)d_in[17];
    const float* bb_hh  = (const float*)d_in[18];
    const float* W_bl   = (const float*)d_in[19];
    const float* b_bl   = (const float*)d_in[20];
    const float* u_pres = (const float*)d_in[21];
    const float* eps_w  = (const float*)d_in[22];
    const float* eps_t  = (const float*)d_in[23];

    // ---- workspace carve-up ----
    char* w = (char*)d_ws;
    auto alloc = [&](size_t bytes) { void* p = (void*)w; w += (bytes + 255) & ~(size_t)255; return p; };

    _Float16* x16    = (_Float16*)alloc((size_t)Bn * KXp * 2);
    _Float16* wihx16 = (_Float16*)alloc((size_t)NGn * KXp * 2);
    _Float16* bwihx16= (_Float16*)alloc((size_t)NGn * KXp * 2);
    _Float16* w2     = (_Float16*)alloc((size_t)NGn * KZp * 2);
    _Float16* bw2    = (_Float16*)alloc((size_t)NGn * KZp * 2);
    _Float16* we1    = (_Float16*)alloc((size_t)NE1p * KOp * 2);
    _Float16* we2    = (_Float16*)alloc((size_t)NE2p * NE1p * 2);
    _Float16* wd1    = (_Float16*)alloc((size_t)ND1p * KWp * 2);
    _Float16* wd2    = (_Float16*)alloc((size_t)ND2p * ND1p * 2);
    _Float16* zh16   = (_Float16*)alloc((size_t)Bn * KZp * 2);
    _Float16* obj16  = (_Float16*)alloc((size_t)Bn * KOp * 2);
    _Float16* e116   = (_Float16*)alloc((size_t)Bn * NE1p * 2);
    _Float16* zw16   = (_Float16*)alloc((size_t)Bn * KWp * 2);
    _Float16* d116   = (_Float16*)alloc((size_t)Bn * ND1p * 2);

    float* bias_l  = (float*)alloc(NGn * 4);
    float* bias_bl = (float*)alloc(NGn * 4);
    float* be1p    = (float*)alloc(NE1p * 4);
    float* be2p    = (float*)alloc(NE2p * 4);
    float* bd1p    = (float*)alloc(ND1p * 4);
    float* bd2p    = (float*)alloc(ND2p * 4);
    float* XW      = (float*)alloc((size_t)Bn * NGn * 4);
    float* XWbl    = (float*)alloc((size_t)Bn * NGn * 4);
    float* gates   = (float*)alloc((size_t)Bn * NGn * 4);
    float* hbuf    = (float*)alloc((size_t)Bn * LHn * 4);
    float* cbuf    = (float*)alloc((size_t)Bn * LHn * 4);
    float* blh     = (float*)alloc((size_t)Bn * LHn * 4);
    float* blc     = (float*)alloc((size_t)Bn * LHn * 4);
    float* zpres   = (float*)alloc((size_t)Bn * 4);
    float* zwhere  = (float*)alloc((size_t)Bn * 3 * 4);
    float* zwhat   = (float*)alloc((size_t)Bn * 50 * 4);
    float* e2      = (float*)alloc((size_t)Bn * NE2p * 4);
    float* dec     = (float*)alloc((size_t)Bn * ND2p * 4);
    float* canvas  = (float*)alloc((size_t)Bn * 2500 * 4);
    float* klb     = (float*)alloc((size_t)Bn * Tn * 4);
    float* bvb     = (float*)alloc((size_t)Bn * Tn * 4);
    float* zplb    = (float*)alloc((size_t)Bn * Tn * 4);
    float* maskb   = (float*)alloc((size_t)Bn * Tn * 4);
    float* llb     = (float*)alloc((size_t)Bn * 4);

    auto gemm = [&](const _Float16* A, int lda, const _Float16* Wt, int ldw,
                    void* out, int ldo, const float* Cadd, int ldc,
                    const float* bias, int N, int K, int act, int outF16) {
        dim3 g(N / 64, Bn / 128);
        wmma_gemm_f16<<<g, 256, 0, stream>>>(A, lda, Wt, ldw, out, ldo, Cadd, ldc, bias, K, act, outF16);
    };

    // ---- one-time staging (runs every launch; deterministic) ----
    cvt_pad_f16_k<<<gb((long)Bn * KXp), 256, 0, stream>>>(x16, x, Bn, 2500, 2500, KXp, (long)Bn * KXp);
    cvt_pad_f16_k<<<gb((long)NGn * KXp), 256, 0, stream>>>(wihx16, W_ih, NGn, 2500, 2554, KXp, (long)NGn * KXp);
    cvt_pad_f16_k<<<gb((long)NGn * KXp), 256, 0, stream>>>(bwihx16, bW_ih, NGn, 2500, 2554, KXp, (long)NGn * KXp);
    build_w2_k<<<gb((long)NGn * KZp), 256, 0, stream>>>(w2, W_ih, W_hh);
    build_w2_k<<<gb((long)NGn * KZp), 256, 0, stream>>>(bw2, bW_ih, bW_hh);
    cvt_pad_f16_k<<<gb((long)NE1p * KOp), 256, 0, stream>>>(we1, W_e1, 200, 400, 400, KOp, (long)NE1p * KOp);
    cvt_pad_f16_k<<<gb((long)NE2p * NE1p), 256, 0, stream>>>(we2, W_e2, 100, 200, 200, NE1p, (long)NE2p * NE1p);
    cvt_pad_f16_k<<<gb((long)ND1p * KWp), 256, 0, stream>>>(wd1, W_d1, 200, 50, 50, KWp, (long)ND1p * KWp);
    cvt_pad_f16_k<<<gb((long)ND2p * ND1p), 256, 0, stream>>>(wd2, W_d2, 2500, 200, 200, ND1p, (long)ND2p * ND1p);
    add2_k<<<gb(NGn), 256, 0, stream>>>(bias_l, b_ih, b_hh, NGn);
    add2_k<<<gb(NGn), 256, 0, stream>>>(bias_bl, bb_ih, bb_hh, NGn);
    pad_bias_k<<<1, 256, 0, stream>>>(be1p, b_e1, 200, NE1p);
    pad_bias_k<<<1, 256, 0, stream>>>(be2p, b_e2, 100, NE2p);
    pad_bias_k<<<1, 256, 0, stream>>>(bd1p, b_d1, 200, ND1p);
    pad_bias_k<<<gb(ND2p), 256, 0, stream>>>(bd2p, b_d2, 2500, ND2p);
    init_state_k<<<gb((long)Bn * 2500), 256, 0, stream>>>(hbuf, cbuf, blh, blc, canvas,
                                                         zpres, zwhere, zwhat, klb, bvb, zplb, maskb);

    // ---- hoisted x-part of both LSTM input GEMMs (x is time-invariant) ----
    gemm(x16, KXp, wihx16, KXp, XW, NGn, nullptr, 0, nullptr, NGn, KXp, 0, 0);
    gemm(x16, KXp, bwihx16, KXp, XWbl, NGn, nullptr, 0, nullptr, NGn, KXp, 0, 0);

    // ---- time loop ----
    for (int t = 0; t < Tn; ++t) {
        build_zh_k<<<gb((long)Bn * KZp), 256, 0, stream>>>(zh16, zwhere, zwhat, zpres, hbuf);

        // main LSTM: gates = XW + zh @ W2^T + (b_ih + b_hh)
        gemm(zh16, KZp, w2, KZp, gates, NGn, XW, NGn, bias_l, NGn, KZp, 0, 0);
        lstm_gate_k<<<gb((long)Bn * LHn), 256, 0, stream>>>(gates, hbuf, cbuf);

        predictor_k<<<gb(Bn), 256, 0, stream>>>(hbuf, W_p, b_p, u_pres, eps_w,
                                                zpres, zwhere, zplb, maskb, klb, t);

        // baseline LSTM (same lstm_in; stop_gradient is a no-op in forward)
        gemm(zh16, KZp, bw2, KZp, gates, NGn, XWbl, NGn, bias_bl, NGn, KZp, 0, 0);
        lstm_gate_k<<<gb((long)Bn * LHn), 256, 0, stream>>>(gates, blh, blc);
        baseline_out_k<<<gb(Bn), 256, 0, stream>>>(blh, W_bl, b_bl, zpres, bvb, t);

        // attention glimpse -> encoder
        obj_sample_k<<<gb((long)Bn * KOp), 256, 0, stream>>>(x, zwhere, obj16);
        gemm(obj16, KOp, we1, KOp, e116, NE1p, nullptr, 0, be1p, NE1p, KOp, 1, 1);   // relu, f16 out
        gemm(e116, NE1p, we2, NE1p, e2, NE2p, nullptr, 0, be2p, NE2p, NE1p, 0, 0);   // f32 out
        zwhat_k<<<gb(Bn), 256, 0, stream>>>(e2, eps_t, zwhat, zw16, klb, t);

        // decoder -> render
        gemm(zw16, KWp, wd1, KWp, d116, ND1p, nullptr, 0, bd1p, ND1p, KWp, 1, 1);    // relu, f16 out
        gemm(d116, ND1p, wd2, ND1p, dec, ND2p, nullptr, 0, bd2p, ND2p, ND1p, 2, 0);  // sigmoid(x-2)
        render_k<<<gb((long)Bn * 2500), 256, 0, stream>>>(dec, zwhere, zpres, canvas);
    }

    // ---- loss reduction ----
    ll_k<<<Bn, 256, 0, stream>>>(x, canvas, llb);
    final_k<<<1, 256, 0, stream>>>(klb, bvb, zplb, maskb, llb, (float*)d_out);

    (void)in_sizes; (void)n_in; (void)out_size; (void)ws_size;
}
